// HyperMixerLayer_44641890075254
// MI455X (gfx1250) — matrix operand order repaired and checked by reading, not compiled, not run on hardware
//
#include <hip/hip_runtime.h>
#include <hip/hip_bf16.h>

// ---------------------------------------------------------------------------
// HyperMixer layer for MI455X (gfx1250): bf16 WMMA GEMM pipeline with
// TDM (tensor_load_to_lds) double-buffered staging.
//   B=2, S=4096, D=1024, DI=4096
// ---------------------------------------------------------------------------

typedef __bf16 bf16_t;
typedef __attribute__((ext_vector_type(16))) __bf16 v16bf;
typedef __attribute__((ext_vector_type(8)))  float  v8f;
typedef __attribute__((ext_vector_type(4)))  unsigned int u32x4;
typedef __attribute__((ext_vector_type(8)))  int i32x8;
typedef __attribute__((ext_vector_type(4)))  int i32x4;

union FragBf { v16bf v; uint4 q[2]; };

#if defined(__HIP_DEVICE_COMPILE__) && __has_builtin(__builtin_amdgcn_tensor_load_to_lds)
#define HM_HAVE_TDM 1
#else
#define HM_HAVE_TDM 0
#endif

#if defined(__HIP_DEVICE_COMPILE__) && __has_builtin(__builtin_amdgcn_s_wait_tensorcnt)
#define HM_WAIT_TENSOR() __builtin_amdgcn_s_wait_tensorcnt((short)0)
#else
#define HM_WAIT_TENSOR() ((void)0)
#endif

__device__ __forceinline__ float gelu_f(float x) {
    // exact erf gelu (reference uses approximate=False)
    return 0.5f * x * (1.0f + erff(x * 0.70710678118654752440f));
}

// ---------------------------------------------------------------------------
// Transpose + f32 -> bf16 convert:  out[c*R + r] = (bf16) in[r*C + c]
// ---------------------------------------------------------------------------
__global__ __launch_bounds__(256)
void hm_transpose_cvt(const float* __restrict__ in, bf16_t* __restrict__ out,
                      int R, int C, long long in_stride, long long out_stride)
{
    __shared__ float tile[32][33];
    const long long ib = (long long)blockIdx.z * in_stride;
    const long long ob = (long long)blockIdx.z * out_stride;
    const int r0 = blockIdx.y * 32, c0 = blockIdx.x * 32;
    const int tx = threadIdx.x & 31, ty = threadIdx.x >> 5;
#pragma unroll
    for (int i = 0; i < 32; i += 8)
        tile[ty + i][tx] = in[ib + (long long)(r0 + ty + i) * C + (c0 + tx)];
    __syncthreads();
#pragma unroll
    for (int i = 0; i < 32; i += 8)
        out[ob + (long long)(c0 + ty + i) * R + (r0 + tx)] = (bf16_t)tile[tx][ty + i];
}

// ---------------------------------------------------------------------------
// Row LayerNorm with optional input-gelu / residual-add epilogue.
// ---------------------------------------------------------------------------
__global__ __launch_bounds__(256)
void hm_layernorm(const float* __restrict__ in, const float* __restrict__ g,
                  const float* __restrict__ bta, const float* __restrict__ res,
                  float* __restrict__ of, bf16_t* __restrict__ ob,
                  int Dn, int do_gelu)
{
    const long long row = blockIdx.x;
    const float* xr = in + row * (long long)Dn;
    const int tid = threadIdx.x;
    const int per = Dn >> 8;
    __shared__ float red[256];

    float local[8];
    float s = 0.f;
    for (int i = 0; i < per; ++i) {
        float v = xr[tid + (i << 8)];
        if (do_gelu) v = gelu_f(v);
        local[i] = v;
        s += v;
    }
    red[tid] = s;
    __syncthreads();
    for (int o = 128; o > 0; o >>= 1) {
        if (tid < o) red[tid] += red[tid + o];
        __syncthreads();
    }
    const float mu = red[0] / (float)Dn;
    __syncthreads();

    float vs = 0.f;
    for (int i = 0; i < per; ++i) {
        float d = local[i] - mu;
        vs += d * d;
    }
    red[tid] = vs;
    __syncthreads();
    for (int o = 128; o > 0; o >>= 1) {
        if (tid < o) red[tid] += red[tid + o];
        __syncthreads();
    }
    const float rstd = rsqrtf(red[0] / (float)Dn + 1e-5f);

    for (int i = 0; i < per; ++i) {
        const int c = tid + (i << 8);
        float y = (local[i] - mu) * rstd * g[c] + bta[c];
        const long long idx = row * (long long)Dn + c;
        if (res) y += res[idx];
        if (of) of[idx] = y;
        if (ob) ob[idx] = (bf16_t)y;
    }
}

// ---------------------------------------------------------------------------
// TDM descriptors (CDNA5 ISA 8.3/8.4). Group1 is loop-invariant per operand
// (tile shape, row stride, LDS pad config) and is built once before the
// K-loop; per K-step we only refresh group0's global address.
// ---------------------------------------------------------------------------
#define BM 128
#define BN 128
#define BK 32
#define LDSP 40  // padded LDS row stride in bf16 elements (80B)

#if HM_HAVE_TDM
__device__ __forceinline__ i32x8 tdm_make_g1(unsigned tile0, unsigned tile1,
                                             unsigned stride0)
{
    i32x8 g1;
    g1[0] = (int)((1u << 16)      // data_size = 2 bytes
                | (1u << 20)      // pad_enable
                | (3u << 22)      // pad_interval: 16 DWORDs (64B)
                | (3u << 25));    // pad_amount:   4 DWORDs (16B)
    g1[1] = (int)((tile0 & 0xffffu) << 16);                               // tensor_dim0 lo16
    g1[2] = (int)(((tile0 >> 16) & 0xffffu) | ((tile1 & 0xffffu) << 16)); // tdim0 hi | tdim1 lo
    g1[3] = (int)(((tile1 >> 16) & 0xffffu) | ((tile0 & 0xffffu) << 16)); // tdim1 hi | tile_dim0
    g1[4] = (int)(tile1 & 0xffffu);                                       // tile_dim1 (tile_dim2=0)
    g1[5] = (int)stride0;                                                 // tensor_dim0_stride[31:0]
    g1[6] = 0;
    g1[7] = 0;
    return g1;
}

__device__ __forceinline__ void tdm_issue(i32x8 g1, unsigned ldsOff,
                                          unsigned long long ga)
{
    u32x4 g0;
    g0[0] = 1u;                                                   // count=1 (user D#)
    g0[1] = ldsOff;                                               // lds_addr
    g0[2] = (unsigned)ga;                                         // global_addr[31:0]
    g0[3] = (unsigned)((ga >> 32) & 0x01ffffffull) | (2u << 30);  // addr[56:32] | type=2
    i32x4 z4 = (i32x4){0, 0, 0, 0};
#if __clang_major__ >= 23
    i32x8 z8 = (i32x8){0, 0, 0, 0, 0, 0, 0, 0};
    __builtin_amdgcn_tensor_load_to_lds(g0, g1, z4, z4, z8, 0);
#else
    __builtin_amdgcn_tensor_load_to_lds(g0, g1, z4, z4, 0);
#endif
}
#endif  // HM_HAVE_TDM

// ---------------------------------------------------------------------------
// General bf16 WMMA GEMM:  C = epilogue( A[M,K] @ Bt[N,K]^T + bias )
//   flags: 1 = gelu, 2 = causal mask (col>row -> 0, skip masked tiles),
//          4 = causal-K truncation (K-loop limited to m0+BM).
// Block: 256 threads / 8 waves (4 row-waves x 2 col-waves).
// Block tile 128x128x32; per-wave 32x64 C-tile = 2x4 WMMA frags.
// ---------------------------------------------------------------------------
__global__ __launch_bounds__(256)
void hm_gemm_wmma(const bf16_t* __restrict__ A, const bf16_t* __restrict__ Bt,
                  const float* __restrict__ bias, const float* __restrict__ residual,
                  float* __restrict__ Cf, bf16_t* __restrict__ Cb,
                  int M, int N, int K,
                  long long strideA, long long strideBt, long long strideC,
                  int lda, int ldbt, int ldc, int flags)
{
    __shared__ __align__(16) bf16_t sA[2][BM * LDSP];
    __shared__ __align__(16) bf16_t sB[2][BN * LDSP];

    const int bz = blockIdx.z;
    A  += (long long)bz * strideA;
    Bt += (long long)bz * strideBt;
    const long long coff = (long long)bz * strideC;

    const int m0   = blockIdx.y * BM;
    const int n0   = blockIdx.x * BN;
    const int tid  = threadIdx.x;
    const int wave = tid >> 5;
    const int lane = tid & 31;
    const int lm   = lane & 15;
    const int lh   = lane >> 4;
    const int wrow = (wave >> 1) * 32;
    const int wcol = (wave & 1) * 64;

    const bool full_mask = ((flags & 2) != 0) && (n0 > m0 + BM - 1);

    v8f acc[2][4];
#pragma unroll
    for (int a = 0; a < 2; ++a)
#pragma unroll
        for (int f = 0; f < 4; ++f)
            acc[a][f] = (v8f){0.f, 0.f, 0.f, 0.f, 0.f, 0.f, 0.f, 0.f};

    if (!full_mask) {
        int kend = K;
        if (flags & 4) { const int lim = m0 + BM; kend = lim < K ? lim : K; }
        const int nk = kend / BK;

        const bf16_t* gA = &A[(long long)m0 * lda];
        const bf16_t* gB = &Bt[(long long)n0 * ldbt];

#if HM_HAVE_TDM
        // hoisted loop-invariant descriptor state
        const i32x8 g1A = tdm_make_g1((unsigned)BK, (unsigned)BM, (unsigned)lda);
        const i32x8 g1B = tdm_make_g1((unsigned)BK, (unsigned)BN, (unsigned)ldbt);
        const unsigned long long gaA = (unsigned long long)gA;
        const unsigned long long gaB = (unsigned long long)gB;
        const unsigned ldsA[2] = { (unsigned)(unsigned long long)(void*)&sA[0][0],
                                   (unsigned)(unsigned long long)(void*)&sA[1][0] };
        const unsigned ldsB[2] = { (unsigned)(unsigned long long)(void*)&sB[0][0],
                                   (unsigned)(unsigned long long)(void*)&sB[1][0] };
        auto stage = [&](int buf, int k) {
            if (tid == 0) {
                tdm_issue(g1A, ldsA[buf], gaA + (unsigned long long)k * 2u);
                tdm_issue(g1B, ldsB[buf], gaB + (unsigned long long)k * 2u);
            }
        };
#else
        auto stage = [&](int buf, int k) {
            for (int c = tid; c < (BM * BK) / 8; c += 256) {
                const int r = c >> 2, col = (c & 3) * 8;
                *(uint4*)&sA[buf][r * LDSP + col] =
                    *(const uint4*)&gA[(long long)r * lda + k + col];
            }
            for (int c = tid; c < (BN * BK) / 8; c += 256) {
                const int r = c >> 2, col = (c & 3) * 8;
                *(uint4*)&sB[buf][r * LDSP + col] =
                    *(const uint4*)&gB[(long long)r * ldbt + k + col];
            }
        };
#endif

        // prologue: stage K-step 0 into buffer 0
        stage(0, 0);
        HM_WAIT_TENSOR();
        __syncthreads();

        int cur = 0;
        for (int i = 0; i < nk; ++i) {
            // async-stage next K-step into the other buffer (overlaps compute)
            if (i + 1 < nk) stage(cur ^ 1, (i + 1) * BK);

            // ---- compute on current buffer ----
            const bf16_t* bufA = &sA[cur][0];
            const bf16_t* bufB = &sB[cur][0];

            FragBf a0, a1;
            {
                const bf16_t* r0 = &bufA[(wrow + lm) * LDSP];
                const bf16_t* r1 = &bufA[(wrow + 16 + lm) * LDSP];
                a0.q[0] = *(const uint4*)(r0 + lh * 8);
                a0.q[1] = *(const uint4*)(r0 + 16 + lh * 8);
                a1.q[0] = *(const uint4*)(r1 + lh * 8);
                a1.q[1] = *(const uint4*)(r1 + 16 + lh * 8);
            }
#pragma unroll
            for (int f = 0; f < 4; ++f) {
                FragBf bfr;
                const bf16_t* br = &bufB[(wcol + f * 16 + lm) * LDSP];
                bfr.q[0] = *(const uint4*)(br + lh * 8);
                bfr.q[1] = *(const uint4*)(br + 16 + lh * 8);
                acc[0][f] = __builtin_amdgcn_wmma_f32_16x16x32_bf16(
                    false, a0.v, false, bfr.v, (short)0, acc[0][f], false, false);
                acc[1][f] = __builtin_amdgcn_wmma_f32_16x16x32_bf16(
                    false, a1.v, false, bfr.v, (short)0, acc[1][f], false, false);
            }

            HM_WAIT_TENSOR();    // issuing wave drains TENSORcnt for next buffer
            __syncthreads();     // publish LDS / protect buffers
            cur ^= 1;
        }
    }

    // ---- epilogue: C layout lane -> (col = lm, rows = lh*8 + r) ----
#pragma unroll
    for (int a = 0; a < 2; ++a) {
#pragma unroll
        for (int f = 0; f < 4; ++f) {
            const int col = n0 + wcol + f * 16 + lm;
#pragma unroll
            for (int r = 0; r < 8; ++r) {
                const int row = m0 + wrow + a * 16 + lh * 8 + r;
                float v = acc[a][f][r];
                if (bias) v += bias[col];
                if (flags & 1) v = gelu_f(v);
                if ((flags & 2) && (col > row)) v = 0.f;
                const long long idx = coff + (long long)row * ldc + col;
                if (residual) v += residual[idx];
                if (Cf) Cf[idx] = v;
                if (Cb) Cb[idx] = (bf16_t)v;
            }
        }
    }
}

// ---------------------------------------------------------------------------
// Host-side launch
// ---------------------------------------------------------------------------
extern "C" void kernel_launch(void* const* d_in, const int* in_sizes, int n_in,
                              void* d_out, int out_size, void* d_ws, size_t ws_size,
                              hipStream_t stream)
{
    (void)in_sizes; (void)n_in; (void)out_size; (void)ws_size;

    const int Bb = 2, S = 4096, D = 1024, DI = 4096;
    const long long SD  = (long long)S * D;
    const long long SDI = (long long)S * DI;
    const long long SS  = (long long)S * S;
    const long long BSD = (long long)Bb * SD;

    const float* x     = (const float*)d_in[0];
    const float* ln_g  = (const float*)d_in[1];
    const float* ln_b  = (const float*)d_in[2];
    const float* h1_w1 = (const float*)d_in[3];
    const float* h1_b1 = (const float*)d_in[4];
    const float* h1_w2 = (const float*)d_in[5];
    const float* h1_b2 = (const float*)d_in[6];
    const float* h2_w1 = (const float*)d_in[7];
    const float* h2_b1 = (const float*)d_in[8];
    const float* h2_w2 = (const float*)d_in[9];
    const float* h2_b2 = (const float*)d_in[10];
    const float* fm_w1 = (const float*)d_in[11];
    const float* fm_b1 = (const float*)d_in[12];
    const float* fm_w2 = (const float*)d_in[13];
    const float* fm_b2 = (const float*)d_in[14];
    float* out = (float*)d_out;

    char* ws = (char*)d_ws;
    size_t off = 0;
    auto take = [&](size_t bytes) -> void* {
        void* p = ws + off;
        off = (off + bytes + 255) & ~(size_t)255;
        return p;
    };
    bf16_t* h1w1T = (bf16_t*)take(sizeof(bf16_t) * (size_t)D * DI);  // [DI,D]
    bf16_t* h1w2T = (bf16_t*)take(sizeof(bf16_t) * (size_t)D * DI);  // [D,DI]
    bf16_t* h2w1T = (bf16_t*)take(sizeof(bf16_t) * (size_t)D * DI);
    bf16_t* h2w2T = (bf16_t*)take(sizeof(bf16_t) * (size_t)D * DI);
    bf16_t* fmw1T = (bf16_t*)take(sizeof(bf16_t) * (size_t)D * DI);
    bf16_t* fmw2T = (bf16_t*)take(sizeof(bf16_t) * (size_t)D * DI);
    float*  n_f   = (float*) take(sizeof(float)  * (size_t)BSD);
    bf16_t* n_bf  = (bf16_t*)take(sizeof(bf16_t) * (size_t)BSD);
    bf16_t* nT_bf = (bf16_t*)take(sizeof(bf16_t) * (size_t)BSD);     // [B][D,S]
    bf16_t* g_bf  = (bf16_t*)take(sizeof(bf16_t) * (size_t)Bb * SDI);
    bf16_t* W1_bf = (bf16_t*)take(sizeof(bf16_t) * (size_t)BSD);
    bf16_t* W2_bf = (bf16_t*)take(sizeof(bf16_t) * (size_t)BSD);
    bf16_t* W3_bf = (bf16_t*)take(sizeof(bf16_t) * (size_t)Bb * SS);
    float*  t_f   = (float*) take(sizeof(float)  * (size_t)BSD);
    float*  x1_f  = (float*) take(sizeof(float)  * (size_t)BSD);
    bf16_t* x1n   = (bf16_t*)take(sizeof(bf16_t) * (size_t)BSD);

    const dim3 blk(256);

    // ---- 1) weights -> bf16, transposed to [N,K] ----
    hm_transpose_cvt<<<dim3(DI/32, D/32, 1), blk, 0, stream>>>(h1_w1, h1w1T, D, DI, 0, 0);
    hm_transpose_cvt<<<dim3(D/32, DI/32, 1), blk, 0, stream>>>(h1_w2, h1w2T, DI, D, 0, 0);
    hm_transpose_cvt<<<dim3(DI/32, D/32, 1), blk, 0, stream>>>(h2_w1, h2w1T, D, DI, 0, 0);
    hm_transpose_cvt<<<dim3(D/32, DI/32, 1), blk, 0, stream>>>(h2_w2, h2w2T, DI, D, 0, 0);
    hm_transpose_cvt<<<dim3(DI/32, D/32, 1), blk, 0, stream>>>(fm_w1, fmw1T, D, DI, 0, 0);
    hm_transpose_cvt<<<dim3(D/32, DI/32, 1), blk, 0, stream>>>(fm_w2, fmw2T, DI, D, 0, 0);

    // ---- 2) n = LN(x);  nT[b] = n[b]^T ----
    hm_layernorm<<<dim3(Bb * S), blk, 0, stream>>>(x, ln_g, ln_b, nullptr,
                                                   n_f, n_bf, D, 0);
    hm_transpose_cvt<<<dim3(D/32, S/32, Bb), blk, 0, stream>>>(n_f, nT_bf, S, D, SD, SD);

    // ---- 3) W1 = mlp(n; h1) ----
    hm_gemm_wmma<<<dim3(DI/BN, (Bb*S)/BM, 1), blk, 0, stream>>>(
        n_bf, h1w1T, h1_b1, nullptr, nullptr, g_bf,
        Bb*S, DI, D, 0, 0, 0, D, D, DI, /*gelu*/1);
    hm_gemm_wmma<<<dim3(D/BN, (Bb*S)/BM, 1), blk, 0, stream>>>(
        g_bf, h1w2T, h1_b2, nullptr, nullptr, W1_bf,
        Bb*S, D, DI, 0, 0, 0, DI, DI, D, 0);

    // ---- 4) W2 = mlp(n; h2) ----
    hm_gemm_wmma<<<dim3(DI/BN, (Bb*S)/BM, 1), blk, 0, stream>>>(
        n_bf, h2w1T, h2_b1, nullptr, nullptr, g_bf,
        Bb*S, DI, D, 0, 0, 0, D, D, DI, 1);
    hm_gemm_wmma<<<dim3(D/BN, (Bb*S)/BM, 1), blk, 0, stream>>>(
        g_bf, h2w2T, h2_b2, nullptr, nullptr, W2_bf,
        Bb*S, D, DI, 0, 0, 0, DI, DI, D, 0);

    // ---- 5) W3 = tril(W1 @ W2^T) (Bt = W2 directly) ----
    hm_gemm_wmma<<<dim3(S/BN, S/BM, Bb), blk, 0, stream>>>(
        W1_bf, W2_bf, nullptr, nullptr, nullptr, W3_bf,
        S, S, D, SD, SD, SS, D, D, S, /*causal*/2);

    // ---- 6) t = W3 @ n (Bt = nT; causal-K truncation) ----
    hm_gemm_wmma<<<dim3(D/BN, S/BM, Bb), blk, 0, stream>>>(
        W3_bf, nT_bf, nullptr, nullptr, t_f, nullptr,
        S, D, S, SS, SD, SD, S, S, D, /*causal_k*/4);

    // ---- 7) x1 = x + LN(gelu(t));  x1n = LN(x1) ----
    hm_layernorm<<<dim3(Bb * S), blk, 0, stream>>>(t_f, ln_g, ln_b, x,
                                                   x1_f, nullptr, D, 1);
    hm_layernorm<<<dim3(Bb * S), blk, 0, stream>>>(x1_f, ln_g, ln_b, nullptr,
                                                   nullptr, x1n, D, 0);

    // ---- 8) out = x + mlp(x1n; fm) ----
    hm_gemm_wmma<<<dim3(DI/BN, (Bb*S)/BM, 1), blk, 0, stream>>>(
        x1n, fmw1T, fm_b1, nullptr, nullptr, g_bf,
        Bb*S, DI, D, 0, 0, 0, D, D, DI, 1);
    hm_gemm_wmma<<<dim3(D/BN, (Bb*S)/BM, 1), blk, 0, stream>>>(
        g_bf, fmw2T, fm_b2, x, out, nullptr,
        Bb*S, D, DI, 0, 0, 0, DI, DI, D, 0);
}
